// DiTAttention_76081050681740
// MI455X (gfx1250) — compile-verified, matrix operand order checked by reading, not run on hardware
//
#include <hip/hip_runtime.h>

typedef __attribute__((ext_vector_type(16))) __bf16        v16bf;
typedef __attribute__((ext_vector_type(8)))  float         v8f;
typedef __attribute__((ext_vector_type(8)))  unsigned int  u32x8;

// ---------- helpers ----------
__device__ __forceinline__ unsigned short f32_bf16(float f) {
  unsigned int u = __float_as_uint(f);
  if ((u & 0x7f800000u) == 0x7f800000u) return (unsigned short)(u >> 16); // inf/nan
  return (unsigned short)((u + 0x7fffu + ((u >> 16) & 1u)) >> 16);        // RNE
}
__device__ __forceinline__ unsigned int pack_bf16x2(float a, float b) {
  return (unsigned int)f32_bf16(a) | ((unsigned int)f32_bf16(b) << 16);
}
// element-pair base K index inside a 32-wide K window for 16-bit A/B fragments
__device__ __forceinline__ int kbe(int v, int hi) {
  return ((v < 4) ? (v * 2) : (v * 2 + 8)) + hi * 8;
}
// generic LDS pointer -> LDS byte offset (flat aperture: low 32 bits give the LDS offset)
__device__ __forceinline__ unsigned lds_off(const void* p) {
  return (unsigned)(unsigned long long)p;
}
// CDNA5 async DMA: 16B global -> LDS, tracked by ASYNCcnt
__device__ __forceinline__ void async_ldsb128(unsigned lds_byte_off, const void* g) {
  asm volatile("global_load_async_to_lds_b128 %0, %1, off"
               :: "v"(lds_byte_off), "v"(g) : "memory");
}
__device__ __forceinline__ void wait_async(int n) {  // n must be a constant
  asm volatile("s_wait_asynccnt %0" :: "n"(n) : "memory");
}

// ---------- 1) f32 -> bf16 convert ----------
__global__ __launch_bounds__(256)
void cvt_bf16_kernel(const float* __restrict__ src, unsigned short* __restrict__ dst, int n) {
  int i = (blockIdx.x * 256 + threadIdx.x) * 4;
  if (i + 4 <= n) {
    float4 f = *reinterpret_cast<const float4*>(src + i);
    uint2 p;
    p.x = pack_bf16x2(f.x, f.y);
    p.y = pack_bf16x2(f.z, f.w);
    *reinterpret_cast<uint2*>(dst + i) = p;
  } else {
    for (int j = i; j < n; ++j) dst[j] = f32_bf16(src[j]);
  }
}

// ---------- 2/5) WMMA GEMM: C[M,N] = A[M,K] * W[N,K]^T + bias[N] ----------
// 256 thr (8 waves), tile 128x128, K-step 32; wave = 64x32 (4x2 wmma tiles)
// double-buffered LDS fed by async global->LDS DMA
__global__ __launch_bounds__(256)
void wmma_gemm_bias(const unsigned short* __restrict__ A,
                    const unsigned short* __restrict__ W,
                    const float* __restrict__ bias,
                    float* __restrict__ C,
                    int M, int N, int K) {
  __shared__ alignas(16) unsigned short ldsA[2][128 * 40];
  __shared__ alignas(16) unsigned short ldsB[2][128 * 40];

  const int tid = threadIdx.x;
  const int lane = tid & 31, lq = lane & 15, hi = lane >> 4;
  const int w = tid >> 5;
  const int waveM = w >> 2;          // 0..1  (64 rows)
  const int waveN = w & 3;           // 0..3  (32 cols)
  const int bm = blockIdx.y * 128;
  const int bn = blockIdx.x * 128;

  v8f acc[4][2];
#pragma unroll
  for (int mf = 0; mf < 4; ++mf)
#pragma unroll
    for (int nf = 0; nf < 2; ++nf)
#pragma unroll
      for (int r = 0; r < 8; ++r) acc[mf][nf][r] = 0.0f;

  const int row  = tid >> 1;   // 0..127
  const int half = tid & 1;
  const int ksteps = K >> 5;

  auto stage = [&](int buf, int kb) {
    const int k0 = kb << 5;
    const unsigned short* ga = A + (unsigned)(bm + row) * K + k0 + half * 16;
    const unsigned la = lds_off(&ldsA[buf][row * 40 + half * 16]);
    async_ldsb128(la,      ga);
    async_ldsb128(la + 16, ga + 8);
    const unsigned short* gw = W + (unsigned)(bn + row) * K + k0 + half * 16;
    const unsigned lb = lds_off(&ldsB[buf][row * 40 + half * 16]);
    async_ldsb128(lb,      gw);
    async_ldsb128(lb + 16, gw + 8);
  };

  stage(0, 0);
  for (int kb = 0; kb < ksteps; ++kb) {
    const int cb = kb & 1;
    if (kb + 1 < ksteps) {        // prefetch next tile, then wait only for current
      stage(cb ^ 1, kb + 1);
      wait_async(4);
    } else {
      wait_async(0);
    }
    __syncthreads();

    v16bf bfrag[2];
#pragma unroll
    for (int nf = 0; nf < 2; ++nf) {
      u32x8 t;
      const int n = waveN * 32 + nf * 16 + lq;
#pragma unroll
      for (int v = 0; v < 8; ++v)
        t[v] = *reinterpret_cast<const unsigned int*>(&ldsB[cb][n * 40 + kbe(v, hi)]);
      bfrag[nf] = __builtin_bit_cast(v16bf, t);
    }
#pragma unroll
    for (int mf = 0; mf < 4; ++mf) {
      u32x8 t;
      const int m = waveM * 64 + mf * 16 + lq;
#pragma unroll
      for (int v = 0; v < 8; ++v)
        t[v] = *reinterpret_cast<const unsigned int*>(&ldsA[cb][m * 40 + kbe(v, hi)]);
      v16bf afrag = __builtin_bit_cast(v16bf, t);
#pragma unroll
      for (int nf = 0; nf < 2; ++nf)
        acc[mf][nf] = __builtin_amdgcn_wmma_f32_16x16x32_bf16(
            false, afrag, false, bfrag[nf], (short)0, acc[mf][nf], false, false);
    }
    __syncthreads();
  }

#pragma unroll
  for (int mf = 0; mf < 4; ++mf) {
    const int mrow = bm + waveM * 64 + mf * 16 + hi * 8;
#pragma unroll
    for (int nf = 0; nf < 2; ++nf) {
      const int n = bn + waveN * 32 + nf * 16 + lq;
      const float bv = bias[n];
#pragma unroll
      for (int r = 0; r < 8; ++r)
        C[(unsigned)(mrow + r) * (unsigned)N + (unsigned)n] = acc[mf][nf][r] + bv;
    }
  }
}

// ---------- 3) RoPE (head 0 only) + split + bf16 pack into (B*H, S, D) ----------
__global__ __launch_bounds__(256)
void rope_split_kernel(const float* __restrict__ qkv,
                       const float* __restrict__ cosb,
                       const float* __restrict__ sinb,
                       unsigned short* __restrict__ qb,
                       unsigned short* __restrict__ kb,
                       unsigned short* __restrict__ vb) {
  int pe = blockIdx.x * 256 + threadIdx.x;
  const int total = 2 * 2048 * 3 * 16 * 32;  // pairs along D
  if (pe >= total) return;
  const int dh  = pe & 31;  pe >>= 5;
  const int h   = pe & 15;  pe >>= 4;
  const int sec = pe % 3;   pe /= 3;
  const int s   = pe & 2047; pe >>= 11;
  const int b   = pe;
  const int d   = dh * 2;

  const unsigned src = (unsigned)(b * 2048 + s) * 3072u + (unsigned)sec * 1024u + h * 64 + d;
  float x1 = qkv[src], x2 = qkv[src + 1];
  float y1 = x1, y2 = x2;
  if (h == 0 && sec < 2) {  // RoPE on q,k head 0; interleaved pairs, cos repeated x2
    const unsigned ci = (unsigned)(b * 2048 + s) * 64u + d;
    const float c = cosb[ci], sn = sinb[ci];
    y1 = x1 * c - x2 * sn;
    y2 = x2 * c + x1 * sn;
  }
  unsigned short* dst = (sec == 0) ? qb : ((sec == 1) ? kb : vb);
  const unsigned di = (unsigned)((b * 16 + h) * 2048 + s) * 64u + d;
  *reinterpret_cast<unsigned int*>(dst + di) = pack_bf16x2(y1, y2);
}

// ---------- 4) flash attention (bf16 WMMA, transposed-score trick, async K tiles) ----------
// grid: (S/128, B*H); block 256 thr = 8 waves; wave handles 16 queries
__global__ __launch_bounds__(256)
void attention_kernel(const unsigned short* __restrict__ Q,
                      const unsigned short* __restrict__ Kc,
                      const unsigned short* __restrict__ V,
                      unsigned short* __restrict__ O) {
  __shared__ alignas(16) unsigned short ldsK[2][32 * 72];   // [key][d], stride 72
  __shared__ alignas(16) unsigned short ldsVt[2][64 * 40];  // [d][key], stride 40 (V^T)

  const int tid = threadIdx.x;
  const int lane = tid & 31, lq = lane & 15, hi = lane >> 4;
  const int w = tid >> 5;
  const int bh = blockIdx.y;
  const int q0 = blockIdx.x * 128 + w * 16;
  const unsigned base = (unsigned)bh * 2048u * 64u;

  // Q^T B-fragments, loaded once (lane = query, elements = d)
  v16bf qf[2];
#pragma unroll
  for (int t = 0; t < 2; ++t) {
    u32x8 tt;
#pragma unroll
    for (int v = 0; v < 8; ++v) {
      const int d = t * 32 + kbe(v, hi);
      tt[v] = *reinterpret_cast<const unsigned int*>(Q + base + (unsigned)(q0 + lq) * 64u + d);
    }
    qf[t] = __builtin_bit_cast(v16bf, tt);
  }

  v8f oacc[4];   // O^T: lane = query, VGPRs+tile = d
#pragma unroll
  for (int a = 0; a < 4; ++a)
#pragma unroll
    for (int r = 0; r < 8; ++r) oacc[a][r] = 0.0f;

  float mrun = -1e30f, lrun = 0.0f;
  const float sc = 0.125f;  // 1/sqrt(64)

  const int skey = tid >> 3;  // 0..31
  const int dseg = tid & 7;   // 0..7

  auto stage = [&](int buf, int kblk) {
    const int kg = kblk * 32 + skey;
    // K tile: direct async DMA to LDS (1 x b128 per thread)
    async_ldsb128(lds_off(&ldsK[buf][skey * 72 + dseg * 8]),
                  Kc + base + (unsigned)kg * 64u + dseg * 8);
    // V tile: transpose through registers
    union { uint4 u; unsigned short s[8]; } cv;
    cv.u = *reinterpret_cast<const uint4*>(V + base + (unsigned)kg * 64u + dseg * 8);
#pragma unroll
    for (int j = 0; j < 8; ++j) ldsVt[buf][(dseg * 8 + j) * 40 + skey] = cv.s[j];
  };

  stage(0, 0);
  for (int kblk = 0; kblk < 64; ++kblk) {
    const int cb = kblk & 1;
    if (kblk + 1 < 64) { stage(cb ^ 1, kblk + 1); wait_async(1); }
    else               { wait_async(0); }
    __syncthreads();

    // S^T = K * Q^T : s0 -> keys 0-15, s1 -> keys 16-31 (lane = query)
    v8f s0, s1;
#pragma unroll
    for (int r = 0; r < 8; ++r) { s0[r] = 0.0f; s1[r] = 0.0f; }
#pragma unroll
    for (int t = 0; t < 2; ++t) {
      u32x8 t0, t1;
#pragma unroll
      for (int v = 0; v < 8; ++v) {
        const int kk = t * 32 + kbe(v, hi);
        t0[v] = *reinterpret_cast<const unsigned int*>(&ldsK[cb][(lq)      * 72 + kk]);
        t1[v] = *reinterpret_cast<const unsigned int*>(&ldsK[cb][(16 + lq) * 72 + kk]);
      }
      v16bf kf0 = __builtin_bit_cast(v16bf, t0);
      v16bf kf1 = __builtin_bit_cast(v16bf, t1);
      s0 = __builtin_amdgcn_wmma_f32_16x16x32_bf16(false, kf0, false, qf[t], (short)0, s0, false, false);
      s1 = __builtin_amdgcn_wmma_f32_16x16x32_bf16(false, kf1, false, qf[t], (short)0, s1, false, false);
    }

    // online softmax: per-query state lane-local; halves merged with one shfl_xor(16)
    float mloc = -1e30f;
#pragma unroll
    for (int r = 0; r < 8; ++r) mloc = fmaxf(mloc, fmaxf(s0[r], s1[r]));
    mloc *= sc;
    const float mblk = fmaxf(mloc, __shfl_xor(mloc, 16));
    const float mnew = fmaxf(mrun, mblk);
    const float corr = __expf(mrun - mnew);
    float p0[8], p1[8], ps = 0.0f;
#pragma unroll
    for (int r = 0; r < 8; ++r) {
      p0[r] = __expf(s0[r] * sc - mnew);
      p1[r] = __expf(s1[r] * sc - mnew);
      ps += p0[r] + p1[r];
    }
    ps += __shfl_xor(ps, 16);
    lrun = lrun * corr + ps;
    mrun = mnew;
#pragma unroll
    for (int a = 0; a < 4; ++a)
#pragma unroll
      for (int r = 0; r < 8; ++r) oacc[a][r] *= corr;

    // S^T accumulator layout == P^T B-fragment layout -> lane-local bf16 pack
    u32x8 pu;
#pragma unroll
    for (int v = 0; v < 4; ++v) pu[v] = pack_bf16x2(p0[2 * v], p0[2 * v + 1]);
#pragma unroll
    for (int v = 4; v < 8; ++v) pu[v] = pack_bf16x2(p1[2 * v - 8], p1[2 * v - 7]);
    const v16bf pf = __builtin_bit_cast(v16bf, pu);

    // O^T += V^T * P^T
#pragma unroll
    for (int a = 0; a < 4; ++a) {
      u32x8 tt;
#pragma unroll
      for (int v = 0; v < 8; ++v)
        tt[v] = *reinterpret_cast<const unsigned int*>(&ldsVt[cb][(a * 16 + lq) * 40 + kbe(v, hi)]);
      v16bf vf = __builtin_bit_cast(v16bf, tt);
      oacc[a] = __builtin_amdgcn_wmma_f32_16x16x32_bf16(false, vf, false, pf, (short)0, oacc[a], false, false);
    }
    __syncthreads();
  }

  // epilogue: normalize, store merged (b, s, h*64+d) bf16
  const float inv = 1.0f / lrun;
  const int b = bh >> 4, h = bh & 15;
  const int srow = q0 + lq;
#pragma unroll
  for (int a = 0; a < 4; ++a)
#pragma unroll
    for (int r = 0; r < 8; ++r) {
      const int d = a * 16 + r + hi * 8;
      O[(unsigned)(b * 2048 + srow) * 1024u + h * 64 + d] = f32_bf16(oacc[a][r] * inv);
    }
}

// ---------- host ----------
extern "C" void kernel_launch(void* const* d_in, const int* in_sizes, int n_in,
                              void* d_out, int out_size, void* d_ws, size_t ws_size,
                              hipStream_t stream) {
  const float* hidden = (const float*)d_in[0];   // (2,2048,1024)
  const float* cosb   = (const float*)d_in[1];   // (2,2048,64)
  const float* sinb   = (const float*)d_in[2];
  const float* qkvw   = (const float*)d_in[3];   // (3072,1024)
  const float* qkvb   = (const float*)d_in[4];   // (3072)
  const float* outw   = (const float*)d_in[5];   // (1024,1024)
  const float* outb   = (const float*)d_in[6];   // (1024)
  float* out = (float*)d_out;                    // (2,2048,1024) f32

  char* ws = (char*)d_ws;
  size_t off = 0;
  auto alloc = [&](size_t bytes) -> void* {
    void* p = ws + off;
    off += (bytes + 255) & ~(size_t)255;
    return p;
  };
  unsigned short* h_bf    = (unsigned short*)alloc((size_t)4194304 * 2);
  unsigned short* qkvw_bf = (unsigned short*)alloc((size_t)3145728 * 2);
  unsigned short* outw_bf = (unsigned short*)alloc((size_t)1048576 * 2);
  float*          qkvf    = (float*)alloc((size_t)4096 * 3072 * 4);
  unsigned short* qb      = (unsigned short*)alloc((size_t)32 * 2048 * 64 * 2);
  unsigned short* kb      = (unsigned short*)alloc((size_t)32 * 2048 * 64 * 2);
  unsigned short* vb      = (unsigned short*)alloc((size_t)32 * 2048 * 64 * 2);
  unsigned short* obf     = h_bf;  // reuse: h_bf dead after QKV GEMM

  cvt_bf16_kernel<<<4096, 256, 0, stream>>>(hidden, h_bf, 4194304);
  cvt_bf16_kernel<<<3072, 256, 0, stream>>>(qkvw, qkvw_bf, 3145728);
  cvt_bf16_kernel<<<1024, 256, 0, stream>>>(outw, outw_bf, 1048576);
  wmma_gemm_bias<<<dim3(24, 32), 256, 0, stream>>>(h_bf, qkvw_bf, qkvb, qkvf, 4096, 3072, 1024);
  rope_split_kernel<<<(2 * 2048 * 3 * 16 * 32 + 255) / 256, 256, 0, stream>>>(qkvf, cosb, sinb, qb, kb, vb);
  attention_kernel<<<dim3(16, 32), 256, 0, stream>>>(qb, kb, vb, obf);
  wmma_gemm_bias<<<dim3(8, 32), 256, 0, stream>>>(obf, outw_bf, outb, out, 4096, 1024, 1024);
}
